// Shift_stcn_block_35751307772329
// MI455X (gfx1250) — compile-verified, matrix operand order checked by reading, not run on hardware
//
#include <hip/hip_runtime.h>
#include <hip/hip_bf16.h>

typedef _Float16 half16 __attribute__((ext_vector_type(16)));
typedef _Float16 half8  __attribute__((ext_vector_type(8)));
typedef float    float8 __attribute__((ext_vector_type(8)));

union H16 { half16 v; half8 h[2]; };

#define TPB 256
#define BB   4
#define CT   256
#define TT   32
#define HH   27
#define HW2  729   /* 27*27 */

// ---------------------------------------------------------------------------
// Source-view mapping: value of the scs-block input tensor at (b, c, hp, wp)
// read directly from x (4,256,32,27,27) with implicit zero padding at h/w==27.
// mode 0: blocks-h   (hp=s,  wp=t*14+s2) -> x[b,cb+c,t,r0+s, c0+s2]
// mode 1: blocks-w   (hp=s2, wp=t*14+s ) -> x[b,cb+c,t,r0+s, c0+hp]
// mode 2: h_s2       (hp=h,  wp=t*28+w ) -> x[b,cb+c,t,hp,   w]
// mode 3: w_s2       (hp=w,  wp=t*28+h ) -> x[b,cb+c,t,h,    hp]
// mode 4: temporal   (hp=t,  wp=hw     ) -> x[b,cb+c,toff+hp, hw]
// ---------------------------------------------------------------------------
__device__ __forceinline__ float src_load(int mode, const float* __restrict__ x,
                                          int b, int c, int hp, int wp,
                                          int cb, int r0, int c0, int toff) {
  int t, h, w;
  switch (mode) {
    case 0: { t = wp / 14; int s2 = wp - t * 14; h = r0 + hp; w = c0 + s2; } break;
    case 1: { t = wp / 14; int s  = wp - t * 14; h = r0 + s;  w = c0 + hp; } break;
    case 2: { t = wp / 28; w = wp - t * 28; h = hp; } break;
    case 3: { t = wp / 28; h = wp - t * 28; w = hp; } break;
    default: {
      size_t idx = (((size_t)b * CT + (cb + c)) * TT + (toff + hp)) * HW2 + wp;
      return x[idx];
    }
  }
  if (h >= HH || w >= HH) return 0.0f;           // padded border of yy
  size_t idx = (((size_t)b * CT + (cb + c)) * TT + t) * HW2 + h * HH + w;
  return x[idx];
}

// ---- per-channel mean / invstd of the (virtual) scs input ------------------
__global__ void k_stats_src(const float* __restrict__ x, float* __restrict__ stats,
                            int mode, int cb, int r0, int c0, int toff,
                            int Hd, int Wd, int N) {
  int c = blockIdx.x;            // 0..63
  int tid = threadIdx.x;
  float s = 0.f, q = 0.f;
  for (int n = tid; n < N; n += TPB) {
    int wp = n % Wd; int rem = n / Wd; int hp = rem % Hd; int b = rem / Hd;
    float v = src_load(mode, x, b, c, hp, wp, cb, r0, c0, toff);
    s += v; q += v * v;
  }
  __shared__ float ls[TPB], lq[TPB];
  ls[tid] = s; lq[tid] = q; __syncthreads();
  for (int off = TPB / 2; off > 0; off >>= 1) {
    if (tid < off) { ls[tid] += ls[tid + off]; lq[tid] += lq[tid + off]; }
    __syncthreads();
  }
  if (tid == 0) {
    float mean = ls[0] / (float)N;
    float var  = lq[0] / (float)N - mean * mean;
    stats[2 * c]     = mean;
    stats[2 * c + 1] = rsqrtf(var + 1e-5f);
  }
}

// ---- BN1 + shift_in, emit f16 activations in [n][64] (K-contiguous) --------
__global__ void k_normalize(const float* __restrict__ x, const float* __restrict__ stats,
                            const float* __restrict__ bnw, const float* __restrict__ bnb,
                            const float* __restrict__ shift, _Float16* __restrict__ X16,
                            int mode, int cb, int r0, int c0, int toff,
                            int Hd, int Wd, int N) {
  int idx = blockIdx.x * TPB + threadIdx.x;
  if (idx >= N * 64) return;
  int c = idx & 63;
  int n = idx >> 6;
  int wp = n % Wd; int rem = n / Wd; int hp = rem % Hd; int b = rem / Hd;
  float pos = fminf(fmaxf((float)wp + shift[c], 0.f), (float)(Wd - 1));
  int i0 = (int)pos;
  int i1 = (i0 + 1 < Wd) ? i0 + 1 : Wd - 1;
  float f = pos - (float)i0;
  float v0 = src_load(mode, x, b, c, hp, i0, cb, r0, c0, toff);
  float v1 = src_load(mode, x, b, c, hp, i1, cb, r0, c0, toff);
  float g  = bnw[c] * stats[2 * c + 1];
  float bb = bnb[c] - stats[2 * c] * g;         // affine commutes with lerp
  float v  = (v0 * (1.f - f) + v1 * f) * g + bb;
  X16[idx] = (_Float16)v;
}

// ---- convert 64x64 weight to f16 -------------------------------------------
__global__ void k_cvt_w(const float* __restrict__ w, _Float16* __restrict__ w16) {
  int i = blockIdx.x * TPB + threadIdx.x;      // 4096 total
  w16[i] = (_Float16)w[i];
}

// ---- 64x64 GEMM via v_wmma_f32_16x16x32_f16, fused bias+ReLU ---------------
// W16: [o][c] f16 row-major (A).  X16: [n][c] f16 (B, K-contiguous).
// Y:   [o][n] f32. 8 waves/block -> 64(o) x 32(n) per block, 2 WMMAs per wave.
__global__ void k_gemm64(const _Float16* __restrict__ W16, const _Float16* __restrict__ X16,
                         const float* __restrict__ bias, float* __restrict__ Y, int N) {
  const int tid  = threadIdx.x;
  const int lane = tid & 31;
  const int wave = tid >> 5;           // 0..7
  const int o0   = (wave & 3) * 16;
  const int n0   = blockIdx.x * 32 + (wave >> 2) * 16;

  const int lh = lane & 15;
  const int kb = (lane >> 4) << 3;     // lanes 0-15: K 0..7 & 16..23; 16-31: +8

  // A operand (weights), K chunks 0..31 and 32..63
  const _Float16* wr = W16 + (o0 + lh) * 64;
  H16 a0, a1;
  a0.h[0] = *(const half8*)(wr + kb);
  a0.h[1] = *(const half8*)(wr + kb + 16);
  a1.h[0] = *(const half8*)(wr + 32 + kb);
  a1.h[1] = *(const half8*)(wr + 32 + kb + 16);

  // B operand (activations), clamp column so EXEC stays all-ones for WMMA
  int ncol = n0 + lh;
  int ncs  = (ncol < N) ? ncol : N - 1;
  const _Float16* xr = X16 + (size_t)ncs * 64;
  H16 b0, b1;
  b0.h[0] = *(const half8*)(xr + kb);
  b0.h[1] = *(const half8*)(xr + kb + 16);
  b1.h[0] = *(const half8*)(xr + 32 + kb);
  b1.h[1] = *(const half8*)(xr + 32 + kb + 16);

  // prefetch next block's activations (global_prefetch_b8)
  int npf = (ncol + 32 < N) ? ncol + 32 : N - 1;
  __builtin_prefetch(X16 + (size_t)npf * 64, 0, 1);

  float8 acc = {0.f, 0.f, 0.f, 0.f, 0.f, 0.f, 0.f, 0.f};
  acc = __builtin_amdgcn_wmma_f32_16x16x32_f16(false, a0.v, false, b0.v,
                                               (short)0, acc, false, false);
  acc = __builtin_amdgcn_wmma_f32_16x16x32_f16(false, a1.v, false, b1.v,
                                               (short)0, acc, false, false);

  const int mbase = o0 + ((lane >> 4) << 3);   // C/D layout: lanes 16-31 -> M+8
  if (ncol < N) {
#pragma unroll
    for (int r = 0; r < 8; ++r) {
      int row = mbase + r;
      float v = acc[r] + bias[row];
      Y[(size_t)row * N + ncol] = (v > 0.f) ? v : 0.f;
    }
  }
}

// ---- per-channel stats of shift_out-interpolated GEMM output ---------------
__global__ void k_stats_y(const float* __restrict__ Y, const float* __restrict__ shift,
                          float* __restrict__ stats, int Hd, int Wd, int N) {
  int c = blockIdx.x;
  int tid = threadIdx.x;
  const float* row = Y + (size_t)c * N;
  float sh = shift[c];
  float s = 0.f, q = 0.f;
  for (int n = tid; n < N; n += TPB) {
    int wp = n % Wd;
    int base = n - wp;
    float pos = fminf(fmaxf((float)wp + sh, 0.f), (float)(Wd - 1));
    int i0 = (int)pos;
    int i1 = (i0 + 1 < Wd) ? i0 + 1 : Wd - 1;
    float f = pos - (float)i0;
    float v = row[base + i0] * (1.f - f) + row[base + i1] * f;
    s += v; q += v * v;
  }
  __shared__ float ls[TPB], lq[TPB];
  ls[tid] = s; lq[tid] = q; __syncthreads();
  for (int off = TPB / 2; off > 0; off >>= 1) {
    if (tid < off) { ls[tid] += ls[tid + off]; lq[tid] += lq[tid + off]; }
    __syncthreads();
  }
  if (tid == 0) {
    float mean = ls[0] / (float)N;
    float var  = lq[0] / (float)N - mean * mean;
    stats[2 * c]     = mean;
    stats[2 * c + 1] = rsqrtf(var + 1e-5f);
  }
}

// ---- shift_out + BN2, scatter through inverse view map ---------------------
// dmode 0/1 -> bs1 (B,128,32,28,28); 2/3 -> bs2; 4 -> temporal (B,64,32,729)
__global__ void k_finalize(const float* __restrict__ Y, const float* __restrict__ stats,
                           const float* __restrict__ bnw, const float* __restrict__ bnb,
                           const float* __restrict__ shift, float* __restrict__ dst,
                           int dmode, int cdst, int toff,
                           int Hd, int Wd, int N, int dh, int dw) {
  int idx = blockIdx.x * TPB + threadIdx.x;
  if (idx >= 64 * N) return;
  int c = idx / N;
  int n = idx - c * N;
  int wp = n % Wd; int rem = n / Wd; int hp = rem % Hd; int b = rem / Hd;

  const float* row = Y + (size_t)c * N;
  int base = n - wp;
  float pos = fminf(fmaxf((float)wp + shift[c], 0.f), (float)(Wd - 1));
  int i0 = (int)pos;
  int i1 = (i0 + 1 < Wd) ? i0 + 1 : Wd - 1;
  float f = pos - (float)i0;
  float v = row[base + i0] * (1.f - f) + row[base + i1] * f;
  float g  = bnw[c] * stats[2 * c + 1];
  float bb = bnb[c] - stats[2 * c] * g;
  v = v * g + bb;

  size_t di;
  switch (dmode) {
    case 0: { int t = wp / 14; int s2 = wp - t * 14;
              di = ((((size_t)b * 128 + cdst + c) * TT + t) * 28 + (dh + hp)) * 28 + (dw + s2); } break;
    case 1: { int t = wp / 14; int s = wp - t * 14;
              di = ((((size_t)b * 128 + cdst + c) * TT + t) * 28 + (dh + s)) * 28 + (dw + hp); } break;
    case 2: { int t = wp / 28; int w = wp - t * 28;
              di = ((((size_t)b * 128 + cdst + c) * TT + t) * 28 + hp) * 28 + w; } break;
    case 3: { int t = wp / 28; int h = wp - t * 28;
              di = ((((size_t)b * 128 + cdst + c) * TT + t) * 28 + h) * 28 + hp; } break;
    default: di = (((size_t)b * 64 + c) * TT + (toff + hp)) * HW2 + wp; break;
  }
  dst[di] = v;
}

// ---- SE global-average over (T,H,W) per (b, ch) ----------------------------
// mode 0: full 128-channel buffer (B,128,32,28,28)
// mode 1: composite (B,128,32,27,27): ch<64 from buf (B,64,32,729), else x group3
__global__ void k_se_mean(const float* __restrict__ buf, const float* __restrict__ x,
                          float* __restrict__ m, int mode) {
  int ch = blockIdx.x;          // 0..127
  int b  = blockIdx.y;          // 0..3
  int tid = threadIdx.x;
  int M;
  const float* p;
  if (mode == 0) {
    M = TT * 28 * 28;
    p = buf + (size_t)(b * 128 + ch) * M;
  } else {
    M = TT * HW2;
    if (ch < 64) p = buf + (size_t)(b * 64 + ch) * M;
    else         p = x + (size_t)(b * CT + 128 + ch) * M;  // x_un channel = 128+ch
  }
  float s = 0.f;
  for (int i = tid; i < M; i += TPB) s += p[i];
  __shared__ float ls[TPB];
  ls[tid] = s; __syncthreads();
  for (int off = TPB / 2; off > 0; off >>= 1) {
    if (tid < off) ls[tid] += ls[tid + off];
    __syncthreads();
  }
  if (tid == 0) m[b * 128 + ch] = ls[0] / (float)M;
}

// ---- SE MLP: 128 -> 8 (relu) -> 128 (sigmoid) ------------------------------
__global__ void k_se_mlp(const float* __restrict__ m, const float* __restrict__ w1,
                         const float* __restrict__ w2, float* __restrict__ att) {
  int b = blockIdx.x;
  int tid = threadIdx.x;        // 128 threads
  __shared__ float mm[128];
  __shared__ float y8[8];
  mm[tid] = m[b * 128 + tid];
  __syncthreads();
  if (tid < 8) {
    float acc = 0.f;
    for (int c = 0; c < 128; ++c) acc += w1[tid * 128 + c] * mm[c];
    y8[tid] = (acc > 0.f) ? acc : 0.f;
  }
  __syncthreads();
  float acc = 0.f;
  for (int j = 0; j < 8; ++j) acc += w2[tid * 8 + j] * y8[j];
  att[b * 128 + tid] = 1.f / (1.f + expf(-acc));
}

// ---- final SE-weighted combine + residual ----------------------------------
__global__ void k_combine(const float* __restrict__ x,
                          const float* __restrict__ bs1, const float* __restrict__ bs2,
                          const float* __restrict__ xt12, const float* __restrict__ xts,
                          const float* __restrict__ att_s1, const float* __restrict__ att_s2,
                          const float* __restrict__ att_t1, const float* __restrict__ att_t2,
                          float* __restrict__ out) {
  size_t idx = (size_t)blockIdx.x * TPB + threadIdx.x;
  const size_t total = (size_t)BB * CT * TT * HW2;
  if (idx >= total) return;
  int hw = (int)(idx % HW2);
  size_t r1 = idx / HW2;
  int t  = (int)(r1 % TT);
  size_t r2 = r1 / TT;
  int ch = (int)(r2 % CT);
  int b  = (int)(r2 / CT);
  float xv = x[idx];
  float res;
  if (ch < 128) {
    int h = hw / HH, w = hw - h * HH;
    size_t bi = ((((size_t)b * 128 + ch) * TT + t) * 28 + h) * 28 + w; // crop [:-1,:-1]
    res = 0.5f * (bs1[bi] * att_s2[b * 128 + ch] + bs2[bi] * att_s1[b * 128 + ch]);
  } else {
    int cc = ch - 128;
    float v12, vs;
    if (cc < 64) {
      size_t ti = (((size_t)b * 64 + cc) * TT + t) * HW2 + hw;
      v12 = xt12[ti]; vs = xts[ti];
    } else {
      v12 = xv; vs = xv;                     // untouched group: value is x itself
    }
    res = 0.5f * (v12 * att_t2[b * 128 + cc] + vs * att_t1[b * 128 + cc]);
  }
  out[idx] = xv + res;
}

// ---------------------------------------------------------------------------
extern "C" void kernel_launch(void* const* d_in, const int* in_sizes, int n_in,
                              void* d_out, int out_size, void* d_ws, size_t ws_size,
                              hipStream_t stream) {
  (void)in_sizes; (void)n_in; (void)out_size; (void)ws_size;
  const float* x = (const float*)d_in[0];
  char* ws = (char*)d_ws;

  float*    bs1  = (float*)(ws + 0);               // 4*128*32*28*28 f32
  float*    bs2  = (float*)(ws + 51380224);
  float*    xt12 = (float*)(ws + 102760448);       // 4*64*32*729 f32
  float*    xts  = (float*)(ws + 126648320);
  float*    Y    = (float*)(ws + 150536192);       // 64 * 100352 f32 (max N)
  _Float16* X16  = (_Float16*)(ws + 176226304);    // 100352 * 64 f16
  _Float16* W16  = (_Float16*)(ws + 189071360);    // 64*64 f16
  float*    st1  = (float*)(ws + 189079552);       // 128 f32
  float*    st2  = (float*)(ws + 189080064);       // 128 f32
  float*    mbuf = (float*)(ws + 189080576);       // 512 f32
  float*    att_s1 = (float*)(ws + 189082624);
  float*    att_s2 = (float*)(ws + 189084672);
  float*    att_t1 = (float*)(ws + 189086720);
  float*    att_t2 = (float*)(ws + 189088768);

  struct D { int mode, cb, r0, c0, toff, Hd, Wd, dmode, cdst, dh, dw, dbuf; };
  // quadrant i -> src (r0,c0) = {(0,0),(0,14),(14,0),(14,14)}, dst (dh,dw) per
  // ba/bb concat = {(0,0),(14,0),(0,14),(14,14)}
  const D ds[13] = {
    {0, 0,   0, 0,  0, 14, 448, 0, 0,  0,  0, 0},   // h1
    {0, 0,   0, 14, 0, 14, 448, 0, 0,  14, 0, 0},   // h2
    {0, 0,  14, 0,  0, 14, 448, 0, 0,  0, 14, 0},   // h3
    {0, 0,  14, 14, 0, 14, 448, 0, 0, 14, 14, 0},   // h4
    {1, 64,  0, 0,  0, 14, 448, 1, 64,  0,  0, 0},  // w1
    {1, 64,  0, 14, 0, 14, 448, 1, 64, 14,  0, 0},  // w2
    {1, 64, 14, 0,  0, 14, 448, 1, 64,  0, 14, 0},  // w3
    {1, 64, 14, 14, 0, 14, 448, 1, 64, 14, 14, 0},  // w4
    {2, 0,   0, 0,  0, 28, 896, 2, 0,   0,  0, 1},  // h_s2
    {3, 64,  0, 0,  0, 28, 896, 3, 64,  0,  0, 1},  // w_s2
    {4, 128, 0, 0,  0, 32, 729, 4, 0,   0,  0, 2},  // t_s1   -> xts
    {4, 128, 0, 0,  0, 16, 729, 4, 0,   0,  0, 3},  // t_s2_1 -> xt12
    {4, 128, 0, 0, 16, 16, 729, 4, 0,   0,  0, 3},  // t_s2_2 -> xt12
  };
  float* dbufs[4] = {bs1, bs2, xts, xt12};

  for (int i = 0; i < 13; ++i) {
    const D& d = ds[i];
    int base = 1 + i * 8;  // bn1_w, bn1_b, shift_in, conv_w, conv_b, shift_out, bn2_w, bn2_b
    const float* bn1w = (const float*)d_in[base + 0];
    const float* bn1b = (const float*)d_in[base + 1];
    const float* sin  = (const float*)d_in[base + 2];
    const float* cw   = (const float*)d_in[base + 3];
    const float* cbv  = (const float*)d_in[base + 4];
    const float* sout = (const float*)d_in[base + 5];
    const float* bn2w = (const float*)d_in[base + 6];
    const float* bn2b = (const float*)d_in[base + 7];
    int N = BB * d.Hd * d.Wd;
    int tot = N * 64;

    k_cvt_w<<<16, TPB, 0, stream>>>(cw, W16);
    k_stats_src<<<64, TPB, 0, stream>>>(x, st1, d.mode, d.cb, d.r0, d.c0, d.toff,
                                        d.Hd, d.Wd, N);
    k_normalize<<<(tot + TPB - 1) / TPB, TPB, 0, stream>>>(x, st1, bn1w, bn1b, sin, X16,
                                                           d.mode, d.cb, d.r0, d.c0,
                                                           d.toff, d.Hd, d.Wd, N);
    k_gemm64<<<(N + 31) / 32, TPB, 0, stream>>>(W16, X16, cbv, Y, N);
    k_stats_y<<<64, TPB, 0, stream>>>(Y, sout, st2, d.Hd, d.Wd, N);
    k_finalize<<<(tot + TPB - 1) / TPB, TPB, 0, stream>>>(Y, st2, bn2w, bn2b, sout,
                                                          dbufs[d.dbuf], d.dmode, d.cdst,
                                                          d.toff, d.Hd, d.Wd, N,
                                                          d.dh, d.dw);
  }

  // SE attentions (se params at indices 105..112: s1_w1,s1_w2,s2_w1,s2_w2,t1_*,t2_*)
  dim3 gse(128, 4);
  k_se_mean<<<gse, TPB, 0, stream>>>(bs1, x, mbuf, 0);
  k_se_mlp<<<4, 128, 0, stream>>>(mbuf, (const float*)d_in[105], (const float*)d_in[106], att_s1);
  k_se_mean<<<gse, TPB, 0, stream>>>(bs2, x, mbuf, 0);
  k_se_mlp<<<4, 128, 0, stream>>>(mbuf, (const float*)d_in[107], (const float*)d_in[108], att_s2);
  k_se_mean<<<gse, TPB, 0, stream>>>(xt12, x, mbuf, 1);
  k_se_mlp<<<4, 128, 0, stream>>>(mbuf, (const float*)d_in[109], (const float*)d_in[110], att_t1);
  k_se_mean<<<gse, TPB, 0, stream>>>(xts, x, mbuf, 1);
  k_se_mlp<<<4, 128, 0, stream>>>(mbuf, (const float*)d_in[111], (const float*)d_in[112], att_t2);

  // Residual combine
  size_t total = (size_t)BB * CT * TT * HW2;
  k_combine<<<(unsigned)((total + TPB - 1) / TPB), TPB, 0, stream>>>(
      x, bs1, bs2, xt12, xts, att_s1, att_s2, att_t1, att_t2, (float*)d_out);
}
